// TNOV2_86002425135419
// MI455X (gfx1250) — matrix operand order repaired and checked by reading, not compiled, not run on hardware
//
#include <hip/hip_runtime.h>
#include <hip/hip_bf16.h>
#include <math.h>

// ---------------- types ----------------
typedef __attribute__((ext_vector_type(16))) __bf16 v16bf;
typedef __attribute__((ext_vector_type(8)))  float  v8f;
typedef __attribute__((ext_vector_type(4))) unsigned int u32x4;
typedef __attribute__((ext_vector_type(8)))  int          i32x8;
typedef __attribute__((ext_vector_type(4)))  int          i32x4;

union Frag16 {
    uint4          q[2];
    v16bf          v;
    unsigned short s[16];
};

// ---------------- dims ----------------
#define B_   8
#define N_   4096
#define DM_  512
#define D1_  1024
#define M2_  8192           // DPB rows (d = r - 4095)
#define NT_  (B_ * N_)      // 32768 tokens
#define RSQRT512 0.04419417382415922f

// epilogue modes
#define MODE_NONE   0
#define MODE_SILU   1
#define MODE_SILU_V 2   // store v as V[((c*8)+b)*4096 + i]
#define MODE_TSTORE 3   // store T[c*8192 + r]
#define MODE_GATE   4   // (acc+bias) * gate[row*Nc+col]
#define MODE_RESID  5   // fp32: acc + bias + resid[row*Nc+col]

__device__ __forceinline__ unsigned short f2bf(float f) {
    unsigned int u = __float_as_uint(f);
    u += 0x7fffu + ((u >> 16) & 1u);
    return (unsigned short)(u >> 16);
}
__device__ __forceinline__ float bf2f(unsigned short s) {
    return __uint_as_float(((unsigned int)s) << 16);
}
__device__ __forceinline__ float silu(float x) {
    return x * (1.f / (1.f + __expf(-x)));
}

// ---------------- fp32 -> bf16 transposed cast: dst[Nc][K] = src[K][Nc] ----------------
__global__ void cast_bf16t_kernel(const float* __restrict__ src,
                                  unsigned short* __restrict__ dst, int K, int Nc) {
    int idx = blockIdx.x * 256 + threadIdx.x;
    if (idx < K * Nc) {
        int k = idx / Nc, n = idx - k * Nc;
        dst[(size_t)n * K + k] = f2bf(src[idx]);
    }
}

// ---------------- SimpleRMSNorm over d=512 (optional ReLU) ----------------
__global__ __launch_bounds__(256) void srms512_kernel(
    const float* __restrict__ inF, const unsigned short* __restrict__ inB,
    unsigned short* __restrict__ out, int relu) {
    __shared__ float red[256];
    int row = blockIdx.x, t = threadIdx.x;
    size_t base = (size_t)row * 512;
    float a, b;
    if (inF) { a = inF[base + t];        b = inF[base + t + 256]; }
    else     { a = bf2f(inB[base + t]);  b = bf2f(inB[base + t + 256]); }
    red[t] = a * a + b * b;
    __syncthreads();
    for (int s = 128; s > 0; s >>= 1) {
        if (t < s) red[t] += red[t + s];
        __syncthreads();
    }
    float norm  = sqrtf(red[0]);
    float scale = 1.f / (norm * RSQRT512 + 1e-8f);
    float va = a * scale, vb = b * scale;
    if (relu) { va = fmaxf(va, 0.f); vb = fmaxf(vb, 0.f); }
    out[base + t] = f2bf(va);
    out[base + t + 256] = f2bf(vb);
}

// ---------------- DPB first layer: scalar pos -> 512 ----------------
__global__ void dpb_first_kernel(const float* __restrict__ Wp,
                                 const float* __restrict__ bp,
                                 unsigned short* __restrict__ out) {
    int idx = blockIdx.x * 256 + threadIdx.x;      // 8192*512 total
    int r = idx >> 9, c = idx & 511;
    float p = (float)(r - 4095);                   // rel pos d in [-4095, 4096]
    out[idx] = f2bf(p * Wp[c] + bp[c]);
}

// ---------------- generic bf16 WMMA GEMM (transposed weights) ----------------
// C[M x Nc] = epilogue( A[M x K] @ W[K x Nc] + bias ), W given transposed WT[Nc][K].
// block: 256 threads = 8 waves, 4(M)x2(N); each wave: 16x32 C => block tile 64x64.
// Software-pipelined: next K-step fragments issued before current WMMAs.
__global__ __launch_bounds__(256) void gemm_wmma_kernel(
    const unsigned short* __restrict__ A, const unsigned short* __restrict__ WT,
    const float* __restrict__ bias, void* __restrict__ outp,
    const unsigned short* __restrict__ gate, const float* __restrict__ resid,
    int M, int K, int Nc, int mode) {
    int lane = threadIdx.x & 31, w = threadIdx.x >> 5;
    int rowbase = blockIdx.y * 64 + (w >> 1) * 16;
    int colbase = blockIdx.x * 64 + (w & 1) * 32;
    int m = lane & 15, g = lane >> 4;

    // per-lane base pointers (all 16B-aligned)
    const unsigned short* ap  = A  + (size_t)(rowbase + m) * K + g * 16;
    const unsigned short* bp0 = WT + (size_t)(colbase + m) * K + g * 16;
    const unsigned short* bp1 = bp0 + (size_t)16 * K;

    Frag16 af, b0f, b1f;
    af.q[0]  = *(const uint4*)(ap);        af.q[1]  = *(const uint4*)(ap + 8);
    b0f.q[0] = *(const uint4*)(bp0);       b0f.q[1] = *(const uint4*)(bp0 + 8);
    b1f.q[0] = *(const uint4*)(bp1);       b1f.q[1] = *(const uint4*)(bp1 + 8);

    v8f acc0 = {}, acc1 = {};
    for (int kb = 0; kb < K; kb += 32) {
        Frag16 an, b0n, b1n;
        int kn = kb + 32;
        if (kn < K) {   // issue next-step loads before consuming current frags
            an.q[0]  = *(const uint4*)(ap + kn);
            an.q[1]  = *(const uint4*)(ap + kn + 8);
            b0n.q[0] = *(const uint4*)(bp0 + kn);
            b0n.q[1] = *(const uint4*)(bp0 + kn + 8);
            b1n.q[0] = *(const uint4*)(bp1 + kn);
            b1n.q[1] = *(const uint4*)(bp1 + kn + 8);
            __builtin_prefetch(ap + kn + 32, 0, 1);
        }
        acc0 = __builtin_amdgcn_wmma_f32_16x16x32_bf16(
            false, af.v, false, b0f.v, (short)0, acc0, false, false);
        acc1 = __builtin_amdgcn_wmma_f32_16x16x32_bf16(
            false, af.v, false, b1f.v, (short)0, acc1, false, false);
        af = an; b0f = b0n; b1f = b1n;
    }

    // C/D map: VGPR r -> row rowbase + r + 8*g, col = colbase + (lane%16) (+16)
#pragma unroll
    for (int t = 0; t < 2; ++t) {
        v8f acc = t ? acc1 : acc0;
        int col = colbase + m + t * 16;
        float bv = bias ? bias[col] : 0.f;
#pragma unroll
        for (int r = 0; r < 8; ++r) {
            int row = rowbase + r + g * 8;
            float v = acc[r] + bv;
            if (mode == MODE_RESID) {
                float* o = (float*)outp;
                size_t idx = (size_t)row * Nc + col;
                o[idx] = v + resid[idx];
            } else {
                unsigned short* o = (unsigned short*)outp;
                if (mode == MODE_SILU || mode == MODE_SILU_V) v = silu(v);
                if (mode == MODE_GATE) v *= bf2f(gate[(size_t)row * Nc + col]);
                size_t idx;
                if (mode == MODE_SILU_V) {
                    int b = row >> 12, i = row & 4095;        // token = b*4096 + i
                    idx = ((size_t)col * 8 + b) * 4096 + i;   // V[c][b][i]
                } else if (mode == MODE_TSTORE) {
                    idx = (size_t)col * 8192 + row;           // T[c][d+4095]
                } else {
                    idx = (size_t)row * Nc + col;
                }
                o[idx] = f2bf(v);
            }
        }
    }
}

// ---------------- Toeplitz apply via WMMA ----------------
// out[b, i, c] = sum_j t_c(i - j) * v[c][b][j]
// grid: x = 128 (2 i-tiles per wave), y = 64 channel groups (8 ch/block)
// Per-channel 16KB t-vector staged into LDS via the Tensor Data Mover
// (unconditional: compile success proves the TDM path is in the binary).
__global__ __launch_bounds__(256) void tno_wmma_kernel(
    const unsigned short* __restrict__ Tker,   // [512][8192] bf16, idx = d + 4095
    const unsigned short* __restrict__ V,      // [512][8][4096] bf16
    unsigned short* __restrict__ Out) {        // [32768][512] bf16 (token-major)
    extern __shared__ unsigned short Tsh[];    // 8 * 8192 ushorts = 128 KB
    int lane = threadIdx.x & 31, w = threadIdx.x >> 5;
    int c  = blockIdx.y * 8 + w;
    int i0 = blockIdx.x * 32;                  // this wave covers rows i0 .. i0+31
    unsigned short* tbase = Tsh + w * 8192;

    // --- TDM: stage Tker[c][0..8191] (16 KB) into this wave's LDS slice ---
    {
        unsigned long long ga =
            (unsigned long long)(Tker + (size_t)__builtin_amdgcn_readfirstlane(c) * 8192);
        unsigned int lo = (unsigned int)__builtin_amdgcn_readfirstlane((int)(ga & 0xffffffffu));
        unsigned int hi = (unsigned int)__builtin_amdgcn_readfirstlane((int)(ga >> 32));
        unsigned int ldsoff =
            (unsigned int)__builtin_amdgcn_readfirstlane(w * 16384);
        // D# group0: count=1 | lds_addr | global_addr[56:0] | type=2
        u32x4 g0 = { 1u, ldsoff, lo, (hi & 0x1ffffffu) | (2u << 30) };
        // D# group1: data_size=2B; tensor_dim0=8192; tensor_dim1=1;
        //            tile_dim0=8192 (1-D tile); tensor_dim0_stride=8192
        i32x8 g1 = { 0x10000, 0x20000000, 0x10000, 0x20000000, 0, 8192, 0, 0 };
        i32x4 z4 = { 0, 0, 0, 0 };
#if defined(__clang_major__) && __clang_major__ >= 23
        i32x8 z8 = { 0, 0, 0, 0, 0, 0, 0, 0 };
        __builtin_amdgcn_tensor_load_to_lds(g0, g1, z4, z4, z8, 0);
#else
        __builtin_amdgcn_tensor_load_to_lds(g0, g1, z4, z4, 0);
#endif
        __builtin_amdgcn_s_wait_tensorcnt(0);
    }

    int m = lane & 15, g = lane >> 4;
    const unsigned short* vp = V + ((size_t)c * 8 + m) * 4096 + g * 16; // valid m<8
    Frag16 bfr, bnx;
    if (m < 8) {
        bfr.q[0] = *(const uint4*)(vp);
        bfr.q[1] = *(const uint4*)(vp + 8);
    } else {
        bfr.q[0] = make_uint4(0, 0, 0, 0);
        bfr.q[1] = make_uint4(0, 0, 0, 0);
        bnx.q[0] = make_uint4(0, 0, 0, 0);
        bnx.q[1] = make_uint4(0, 0, 0, 0);
    }

    v8f acc0 = {}, acc1 = {};
    for (int j0 = 0; j0 < 4096; j0 += 32) {
        // pipeline: issue next B tile before consuming current fragments
        int jn = j0 + 32;
        if (m < 8 && jn < 4096) {
            bnx.q[0] = *(const uint4*)(vp + jn);       // K = g*16 + 0..7
            bnx.q[1] = *(const uint4*)(vp + jn + 8);   // K = g*16 + 8..15
        }
        // --- A fragments: two Toeplitz tiles from the LDS t-vector ---
        Frag16 a0, a1;
        int bidx = i0 + m - j0 + 4095;
#pragma unroll
        for (int e = 0; e < 16; ++e) {
            int K = e + g * 16;   // contiguous K = g*16 + e (matches B)
            a0.s[e] = tbase[bidx - K];
            a1.s[e] = tbase[bidx + 16 - K];
        }
        acc0 = __builtin_amdgcn_wmma_f32_16x16x32_bf16(
            false, a0.v, false, bfr.v, (short)0, acc0, false, false);
        acc1 = __builtin_amdgcn_wmma_f32_16x16x32_bf16(
            false, a1.v, false, bfr.v, (short)0, acc1, false, false);
        bfr = bnx;
    }
    // --- store: row token = b*4096 + i, col = channel ---
    if (m < 8) {
#pragma unroll
        for (int r = 0; r < 8; ++r) {
            int i = i0 + r + g * 8;
            Out[((size_t)m * 4096 + i) * 512 + c]      = f2bf(acc0[r]);
            Out[((size_t)m * 4096 + i + 16) * 512 + c] = f2bf(acc1[r]);
        }
    }
}

// ---------------- host launch ----------------
extern "C" void kernel_launch(void* const* d_in, const int* in_sizes, int n_in,
                              void* d_out, int out_size, void* d_ws, size_t ws_size,
                              hipStream_t stream) {
    (void)in_sizes; (void)n_in; (void)out_size; (void)ws_size;
    const float* x   = (const float*)d_in[0];
    const float* Wu  = (const float*)d_in[1];  const float* bu  = (const float*)d_in[2];
    const float* Wv  = (const float*)d_in[3];  const float* bv  = (const float*)d_in[4];
    const float* Wo1 = (const float*)d_in[5];  const float* bo1 = (const float*)d_in[6];
    const float* Wo2 = (const float*)d_in[7];  const float* bo2 = (const float*)d_in[8];
    const float* dWp = (const float*)d_in[9];  const float* dbp = (const float*)d_in[10];
    const float* dW1 = (const float*)d_in[11]; const float* db1 = (const float*)d_in[12];
    const float* dW2 = (const float*)d_in[13]; const float* db2 = (const float*)d_in[14];
    const float* dW3 = (const float*)d_in[15]; const float* db3 = (const float*)d_in[16];

    char* ws = (char*)d_ws;
    size_t off = 0;
    auto take = [&](size_t bytes) { char* p = ws + off; off += bytes; return p; };
    unsigned short* XN  = (unsigned short*)take((size_t)NT_ * DM_ * 2);   // 33.5 MB
    unsigned short* U   = (unsigned short*)take((size_t)NT_ * D1_ * 2);   // 67 MB
    unsigned short* Vb  = (unsigned short*)take((size_t)512 * 8 * N_ * 2);// 33.5 MB
    unsigned short* D0  = (unsigned short*)take((size_t)M2_ * 512 * 2);   // 8.4 MB
    unsigned short* D1b = (unsigned short*)take((size_t)M2_ * 512 * 2);   // 8.4 MB
    unsigned short* TK  = (unsigned short*)take((size_t)512 * 8192 * 2);  // 8.4 MB
    unsigned short* TO  = (unsigned short*)take((size_t)NT_ * 512 * 2);   // 33.5 MB
    unsigned short* G   = (unsigned short*)take((size_t)NT_ * D1_ * 2);   // 67 MB
    unsigned short* WuT  = (unsigned short*)take((size_t)512 * 1024 * 2); // transposed
    unsigned short* WvT  = (unsigned short*)take((size_t)512 * 512 * 2);
    unsigned short* Wo1T = (unsigned short*)take((size_t)512 * 1024 * 2);
    unsigned short* Wo2T = (unsigned short*)take((size_t)1024 * 512 * 2);
    unsigned short* W1T  = (unsigned short*)take((size_t)512 * 512 * 2);
    unsigned short* W2T  = (unsigned short*)take((size_t)512 * 512 * 2);
    unsigned short* W3T  = (unsigned short*)take((size_t)512 * 512 * 2);

    auto castT = [&](const float* s, unsigned short* d, int K, int Nc) {
        cast_bf16t_kernel<<<(K * Nc + 255) / 256, 256, 0, stream>>>(s, d, K, Nc);
    };
    castT(Wu,  WuT,  512, 1024);
    castT(Wv,  WvT,  512, 512);
    castT(Wo1, Wo1T, 512, 1024);
    castT(Wo2, Wo2T, 1024, 512);
    castT(dW1, W1T,  512, 512);
    castT(dW2, W2T,  512, 512);
    castT(dW3, W3T,  512, 512);

    // pre-norm
    srms512_kernel<<<NT_, 256, 0, stream>>>(x, nullptr, XN, 0);

    // u = silu(xn @ Wu + bu)  [32768 x 1024]
    gemm_wmma_kernel<<<dim3(1024 / 64, NT_ / 64), 256, 0, stream>>>(
        XN, WuT, bu, U, nullptr, nullptr, NT_, 512, 1024, MODE_SILU);
    // v = silu(xn @ Wv + bv), stored channel/batch-major [c][b][i]
    gemm_wmma_kernel<<<dim3(512 / 64, NT_ / 64), 256, 0, stream>>>(
        XN, WvT, bv, Vb, nullptr, nullptr, NT_, 512, 512, MODE_SILU_V);

    // DPB MLP over rel positions d = r - 4095
    dpb_first_kernel<<<(M2_ * 512) / 256, 256, 0, stream>>>(dWp, dbp, D0);
    srms512_kernel<<<M2_, 256, 0, stream>>>(nullptr, D0, D1b, 1);
    gemm_wmma_kernel<<<dim3(512 / 64, M2_ / 64), 256, 0, stream>>>(
        D1b, W1T, db1, D0, nullptr, nullptr, M2_, 512, 512, MODE_NONE);
    srms512_kernel<<<M2_, 256, 0, stream>>>(nullptr, D0, D1b, 1);
    gemm_wmma_kernel<<<dim3(512 / 64, M2_ / 64), 256, 0, stream>>>(
        D1b, W2T, db2, D0, nullptr, nullptr, M2_, 512, 512, MODE_NONE);
    srms512_kernel<<<M2_, 256, 0, stream>>>(nullptr, D0, D1b, 1);
    gemm_wmma_kernel<<<dim3(512 / 64, M2_ / 64), 256, 0, stream>>>(
        D1b, W3T, db3, TK, nullptr, nullptr, M2_, 512, 512, MODE_TSTORE);

    // Toeplitz apply (direct, tensor cores): 128 KB dynamic LDS (8 ch x 16 KB)
    tno_wmma_kernel<<<dim3(N_ / 32, 512 / 8), 256, 131072, stream>>>(TK, Vb, TO);

    // gated o1: (TO @ Wo1 + bo1) * u
    gemm_wmma_kernel<<<dim3(1024 / 64, NT_ / 64), 256, 0, stream>>>(
        TO, Wo1T, bo1, G, U, nullptr, NT_, 512, 1024, MODE_GATE);
    // o2 + residual -> fp32 output
    gemm_wmma_kernel<<<dim3(512 / 64, NT_ / 64), 256, 0, stream>>>(
        G, Wo2T, bo2, d_out, nullptr, x, NT_, 1024, 512, MODE_RESID);
}
